// DyGraphTransformer_87342454931906
// MI455X (gfx1250) — matrix operand order repaired and checked by reading, not compiled, not run on hardware
//
#include <hip/hip_runtime.h>
#include <hip/hip_bf16.h>
#include <math.h>

// Problem constants (match reference)
#define NND 512   // nodes
#define FD  256   // features
#define HD  256   // hidden
#define NHD 8     // heads
#define LD  6     // layers
#define DK  32    // head dim

typedef __attribute__((ext_vector_type(16))) __bf16 v16bf;
typedef __attribute__((ext_vector_type(8)))  float  v8f;
typedef int vec4i __attribute__((vector_size(16)));   // matches builtin param type

union Frag16 { uint4 q[2]; v16bf b; unsigned short s[16]; };

#define AS1 __attribute__((address_space(1)))
#define AS3 __attribute__((address_space(3)))

#if defined(__has_builtin)
#if __has_builtin(__builtin_amdgcn_global_load_async_to_lds_b128) && \
    __has_builtin(__builtin_amdgcn_s_wait_asynccnt)
#define HAVE_ASYNC_LDS 1
#endif
#endif
#ifndef HAVE_ASYNC_LDS
#define HAVE_ASYNC_LDS 0
#endif

__device__ __forceinline__ unsigned short f32_to_bf16(float f) {
  unsigned int u = __float_as_uint(f);
  return (unsigned short)((u + 0x7FFFu + ((u >> 16) & 1u)) >> 16);
}
__device__ __forceinline__ float gelu_exact(float x) {
  return 0.5f * x * (1.0f + erff(x * 0.70710678118654752f));
}

// 16-byte global -> LDS copy: async (ASYNCcnt / CDNA5 path) or fallback
__device__ __forceinline__ void copy16_g2l(const unsigned short* g, unsigned short* l) {
#if HAVE_ASYNC_LDS
  __builtin_amdgcn_global_load_async_to_lds_b128(
      (AS1 vec4i*)(unsigned short*)g, (AS3 vec4i*)l, 0, 0);
#else
  *reinterpret_cast<uint4*>(l) = *reinterpret_cast<const uint4*>(g);
#endif
}
__device__ __forceinline__ void wait_g2l() {
#if HAVE_ASYNC_LDS
  __builtin_amdgcn_s_wait_asynccnt(0);
#endif
}

// -------------------------------------------------------------------------
// Small tables: maxnorm(embedding row) @ W_proj  -> tab[160][8]
// rows 0..31  = edge_emb (W_ee) ; rows 32..159 = edge_dist_emb (W_ed)
// -------------------------------------------------------------------------
__global__ __launch_bounds__(256) void k_tables(
    const float* __restrict__ edge_emb, const float* __restrict__ edge_dist_emb,
    const float* __restrict__ W_ee, const float* __restrict__ W_ed,
    float* __restrict__ tab) {
  const int r = blockIdx.x;            // 0..159
  const int t = threadIdx.x;           // 0..255
  const float* row = (r < 32) ? (edge_emb + r * HD) : (edge_dist_emb + (r - 32) * HD);
  const float* W   = (r < 32) ? W_ee : W_ed;   // [H][NH]
  __shared__ float red[256];
  float xv = row[t];
  red[t] = xv * xv;
  __syncthreads();
  for (int s = 128; s > 0; s >>= 1) { if (t < s) red[t] += red[t + s]; __syncthreads(); }
  float nrm = sqrtf(red[0]);
  float scale = (nrm > 1.0f) ? (1.0f / (nrm + 1e-7f)) : 1.0f;
  __syncthreads();
  const int c = t & 7, seg = t >> 3;
  float p = 0.f;
  #pragma unroll
  for (int j = 0; j < 8; ++j) {
    int kk = seg * 8 + j;
    p += row[kk] * W[kk * NHD + c];
  }
  red[t] = p;
  __syncthreads();
  if (t < 8) {
    float s = 0.f;
    for (int g = 0; g < 32; ++g) s += red[g * 8 + t];
    tab[r * 8 + t] = s * scale;
  }
}

// -------------------------------------------------------------------------
// attn_bias[h][i][j] = 0.5*(tab_ee[e0]+tab_ee[e1]) + b_ee + tab_ed[d] + b_ed
// -------------------------------------------------------------------------
__global__ __launch_bounds__(256) void k_attn_bias(
    const int* __restrict__ ee, const int* __restrict__ ed,
    const float* __restrict__ tab, const float* __restrict__ b_ee,
    const float* __restrict__ b_ed, float* __restrict__ bias) {
  const int idx = blockIdx.x * 256 + threadIdx.x;        // over N*N
  const int e0 = ee[idx * 2 + 0], e1 = ee[idx * 2 + 1], dd = ed[idx];
  const float* t0 = tab + e0 * 8;
  const float* t1 = tab + e1 * 8;
  const float* td = tab + (32 + dd) * 8;
  #pragma unroll
  for (int hh = 0; hh < NHD; ++hh) {
    bias[(size_t)hh * NND * NND + idx] =
        0.5f * (t0[hh] + t1[hh]) + b_ee[hh] + td[hh] + b_ed[hh];
  }
}

__global__ __launch_bounds__(256) void k_f32_to_bf16(
    const float* __restrict__ in, unsigned short* __restrict__ out, int n) {
  int i = blockIdx.x * 256 + threadIdx.x;
  if (i < n) out[i] = f32_to_bf16(in[i]);
}

// Convert + transpose weight layers: in f32 [L][K][NN] -> out bf16 [L][NN][K]
__global__ __launch_bounds__(256) void k_convt(
    const float* __restrict__ in, unsigned short* __restrict__ out, int K, int NN) {
  const size_t base = (size_t)blockIdx.y * K * NN;
  const int idx = blockIdx.x * 256 + threadIdx.x;     // coalesced read index
  const int k = idx / NN, n = idx % NN;
  out[base + (size_t)n * K + k] = f32_to_bf16(in[base + idx]);
}

// -------------------------------------------------------------------------
// LayerNorm over H=256, one row per block, bf16 output
// -------------------------------------------------------------------------
__global__ __launch_bounds__(256) void k_layernorm(
    const float* __restrict__ h, const float* __restrict__ s,
    const float* __restrict__ b, unsigned short* __restrict__ y) {
  const int row = blockIdx.x, t = threadIdx.x;
  __shared__ float red[256];
  float xv = h[row * HD + t];
  red[t] = xv;
  __syncthreads();
  for (int st = 128; st > 0; st >>= 1) { if (t < st) red[t] += red[t + st]; __syncthreads(); }
  float mean = red[0] * (1.0f / HD);
  __syncthreads();
  float d = xv - mean;
  red[t] = d * d;
  __syncthreads();
  for (int st = 128; st > 0; st >>= 1) { if (t < st) red[t] += red[t + st]; __syncthreads(); }
  float inv = rsqrtf(red[0] * (1.0f / HD) + 1e-5f);
  y[row * HD + t] = f32_to_bf16(d * inv * s[t] + b[t]);
}

// -------------------------------------------------------------------------
// GEMM: C = act((A_bf16[M,K] @ Bt_bf16[NN,K]^T + bias) * alpha)
//   Bt is the pre-transposed bf16 weight (row n holds K contiguous values).
// block = 128 threads (4 waves), 64x64 tile; wave = 32x32 (2x2 WMMA).
// Double-buffered LDS; tiles staged with async global->LDS copies.
// ACT: 0 none, 1 exact GELU
// OUTMODE: 0 store f32, 1 add into f32 (residual), 2 store bf16,
//          3 store bf16 transposed (C^T, leading dim M)
// -------------------------------------------------------------------------
template<int ACT, int OUTMODE>
__global__ __launch_bounds__(128) void k_gemm_bf16(
    const unsigned short* __restrict__ A, const unsigned short* __restrict__ Bt,
    const float* __restrict__ bias, float alpha,
    float* __restrict__ Cf, unsigned short* __restrict__ Cb,
    int M, int K, int NN) {
  __shared__ __align__(16) unsigned short As[2][64][40];   // [m][k], padded
  __shared__ __align__(16) unsigned short Bs[2][64][40];   // [n][k], padded
  const int t = threadIdx.x;
  const int lane = t & 31, w = t >> 5;
  const int wm = (w >> 1) * 32, wn = (w & 1) * 32;
  const int m0 = blockIdx.x * 64, n0 = blockIdx.y * 64;
  v8f acc[2][2] = {};
  const int nk = K / 32;
  const int arow = lane & 15, kb8 = (lane >> 4) * 8;
  const int bcol = lane & 15, klo = (lane >> 4) * 16;

  // tile stager: A 64x32 (2 chunks/thread) + B 64x32 (2 chunks/thread)
  auto stage = [&](int k0, int buf) {
    const int rowA = t >> 1, cgA = (t & 1) * 16;
    const unsigned short* ga = A + (m0 + rowA) * K + k0 + cgA;
    copy16_g2l(ga,     &As[buf][rowA][cgA]);
    copy16_g2l(ga + 8, &As[buf][rowA][cgA + 8]);
    #pragma unroll
    for (int rep = 0; rep < 2; ++rep) {
      int c = rep * 128 + t;                 // 256 chunks of 8 halfwords
      int nB = c >> 2, kc = (c & 3) * 8;
      copy16_g2l(Bt + (n0 + nB) * K + k0 + kc, &Bs[buf][nB][kc]);
    }
  };

  stage(0, 0);
  int buf = 0;
  for (int kk = 0; kk < nk; ++kk) {
    wait_g2l();
    __syncthreads();
    if (kk + 1 < nk) stage((kk + 1) * 32, buf ^ 1);
    Frag16 a[2], bb[2];
    #pragma unroll
    for (int mi = 0; mi < 2; ++mi) {
      a[mi].q[0] = *reinterpret_cast<const uint4*>(&As[buf][wm + mi * 16 + arow][kb8]);
      a[mi].q[1] = *reinterpret_cast<const uint4*>(&As[buf][wm + mi * 16 + arow][kb8 + 16]);
    }
    #pragma unroll
    for (int ni = 0; ni < 2; ++ni) {
      const uint4* p = reinterpret_cast<const uint4*>(&Bs[buf][wn + ni * 16 + bcol][kb8]);
      bb[ni].q[0] = p[0];
      bb[ni].q[1] = p[2];
    }
    #pragma unroll
    for (int mi = 0; mi < 2; ++mi)
      #pragma unroll
      for (int ni = 0; ni < 2; ++ni)
        acc[mi][ni] = __builtin_amdgcn_wmma_f32_16x16x32_bf16(
            false, a[mi].b, false, bb[ni].b, (short)0, acc[mi][ni], false, false);
    buf ^= 1;
  }
  const int nlo = lane & 15, mhi = (lane >> 4) * 8;
  #pragma unroll
  for (int mi = 0; mi < 2; ++mi)
    #pragma unroll
    for (int ni = 0; ni < 2; ++ni)
      #pragma unroll
      for (int i = 0; i < 8; ++i) {
        int m = m0 + wm + mi * 16 + mhi + i;
        int n = n0 + wn + ni * 16 + nlo;
        float v = (acc[mi][ni][i] + bias[n]) * alpha;
        if (ACT == 1) v = gelu_exact(v);
        if (OUTMODE == 0)      Cf[m * NN + n] = v;
        else if (OUTMODE == 1) Cf[m * NN + n] += v;
        else if (OUTMODE == 2) Cb[m * NN + n] = f32_to_bf16(v);
        else                   Cb[(size_t)n * M + m] = f32_to_bf16(v);  // C^T
      }
}

// -------------------------------------------------------------------------
// Fused attention: one wave per (head, 16-row query tile).
// scores (16x512) via 32 WMMAs + bias, exact softmax (unnormalized exp in
// bf16; row sums folded into PV epilogue), PV via 32 WMMAs reading V^T.
// -------------------------------------------------------------------------
__global__ __launch_bounds__(32) void k_attention(
    const unsigned short* __restrict__ q, const unsigned short* __restrict__ k,
    const unsigned short* __restrict__ vt,   // V^T : [HD][NND] bf16
    const float* __restrict__ bias, unsigned short* __restrict__ o) {
  const int hh = blockIdx.x;     // head
  const int qt = blockIdx.y;     // query tile (16 rows)
  const int lane = threadIdx.x;
  __shared__ __align__(16) float          s_s[16][NND];
  __shared__ __align__(16) unsigned short s_p[16][NND + 8];
  __shared__ float s_sum[16];

  const int nlo = lane & 15, hi = lane >> 4;
  const int kb8 = hi * 8, klo = hi * 16;

  Frag16 aq;   // Q fragment, reused for all 32 key tiles
  {
    const unsigned short* base = q + (qt * 16 + nlo) * HD + hh * DK + kb8;
    aq.q[0] = *reinterpret_cast<const uint4*>(base);
    aq.q[1] = *reinterpret_cast<const uint4*>(base + 16);
  }
  for (int kt = 0; kt < NND / 16; ++kt) {
    Frag16 bk;
    const unsigned short* base = k + (kt * 16 + nlo) * HD + hh * DK + kb8;
    bk.q[0] = *reinterpret_cast<const uint4*>(base);
    bk.q[1] = *reinterpret_cast<const uint4*>(base + 16);
    v8f sc = {};
    sc = __builtin_amdgcn_wmma_f32_16x16x32_bf16(false, aq.b, false, bk.b,
                                                 (short)0, sc, false, false);
    #pragma unroll
    for (int i = 0; i < 8; ++i) {
      int m = hi * 8 + i;
      int qrow = qt * 16 + m;
      int kcol = kt * 16 + nlo;
      s_s[m][kcol] = sc[i] + bias[((size_t)hh * NND + qrow) * NND + kcol];
    }
  }
  __syncthreads();
  for (int r = 0; r < 16; ++r) {
    float mx = -3.0e38f;
    for (int c = lane; c < NND; c += 32) mx = fmaxf(mx, s_s[r][c]);
    for (int off = 16; off > 0; off >>= 1) mx = fmaxf(mx, __shfl_xor(mx, off, 32));
    float sum = 0.f;
    for (int c = lane; c < NND; c += 32) {
      float e = __expf(s_s[r][c] - mx);
      s_p[r][c] = f32_to_bf16(e);
      sum += e;
    }
    for (int off = 16; off > 0; off >>= 1) sum += __shfl_xor(sum, off, 32);
    if (lane == 0) s_sum[r] = sum;
  }
  __syncthreads();
  v8f acc[2] = {};
  for (int kb = 0; kb < NND / 32; ++kb) {
    Frag16 ap;
    const unsigned short* pb = &s_p[nlo][kb * 32 + kb8];
    ap.q[0] = *reinterpret_cast<const uint4*>(pb);
    ap.q[1] = *reinterpret_cast<const uint4*>(pb + 16);
    #pragma unroll
    for (int nt = 0; nt < 2; ++nt) {
      Frag16 bv;   // e[i] = V[kb*32+kb8+i (+8 for i>=8)][hh*DK+nt*16+col] from Vt rows
      const unsigned short* vbase = vt + (size_t)(hh * DK + nt * 16 + nlo) * NND + kb * 32 + kb8;
      bv.q[0] = *reinterpret_cast<const uint4*>(vbase);
      bv.q[1] = *reinterpret_cast<const uint4*>(vbase + 16);
      acc[nt] = __builtin_amdgcn_wmma_f32_16x16x32_bf16(
          false, ap.b, false, bv.b, (short)0, acc[nt], false, false);
    }
  }
  #pragma unroll
  for (int nt = 0; nt < 2; ++nt)
    #pragma unroll
    for (int i = 0; i < 8; ++i) {
      int m = hi * 8 + i;
      o[(qt * 16 + m) * HD + hh * DK + nt * 16 + nlo] =
          f32_to_bf16(acc[nt][i] / s_sum[m]);
    }
}

// -------------------------------------------------------------------------
extern "C" void kernel_launch(void* const* d_in, const int* in_sizes, int n_in,
                              void* d_out, int out_size, void* d_ws, size_t ws_size,
                              hipStream_t stream) {
  (void)in_sizes; (void)n_in; (void)out_size; (void)ws_size;
  const float* x        = (const float*)d_in[0];
  const int*   ee       = (const int*)d_in[2];
  const int*   ed       = (const int*)d_in[3];
  const float* W_feat   = (const float*)d_in[7];
  const float* b_feat   = (const float*)d_in[8];
  const float* edge_emb = (const float*)d_in[9];
  const float* edge_dist_emb = (const float*)d_in[10];
  const float* W_ee     = (const float*)d_in[11];
  const float* b_ee     = (const float*)d_in[12];
  const float* W_ed     = (const float*)d_in[13];
  const float* b_ed     = (const float*)d_in[14];
  const float* ln1_s    = (const float*)d_in[15];
  const float* ln1_b    = (const float*)d_in[16];
  const float* Wq       = (const float*)d_in[17];
  const float* bq       = (const float*)d_in[18];
  const float* Wk       = (const float*)d_in[19];
  const float* bk       = (const float*)d_in[20];
  const float* Wv       = (const float*)d_in[21];
  const float* bv       = (const float*)d_in[22];
  const float* Wo       = (const float*)d_in[23];
  const float* bo       = (const float*)d_in[24];
  const float* ln2_s    = (const float*)d_in[25];
  const float* ln2_b    = (const float*)d_in[26];
  const float* W1       = (const float*)d_in[27];
  const float* b1       = (const float*)d_in[28];
  const float* W2       = (const float*)d_in[29];
  const float* b2       = (const float*)d_in[30];

  char* ws = (char*)d_ws;
  size_t off = 0;
  auto wsalloc = [&](size_t bytes) -> char* {
    char* p = ws + off;
    off += (bytes + 255) & ~(size_t)255;
    return p;
  };
  const size_t WSZ = (size_t)HD * HD;          // one weight layer
  float* tab   = (float*)wsalloc(160 * 8 * sizeof(float));
  float* abias = (float*)wsalloc((size_t)NHD * NND * NND * sizeof(float));
  float* hbuf  = (float*)wsalloc((size_t)NND * HD * sizeof(float));
  unsigned short* xb   = (unsigned short*)wsalloc((size_t)NND * HD * 2);
  unsigned short* yb   = (unsigned short*)wsalloc((size_t)NND * HD * 2);
  unsigned short* qb   = (unsigned short*)wsalloc((size_t)NND * HD * 2);
  unsigned short* kb16 = (unsigned short*)wsalloc((size_t)NND * HD * 2);
  unsigned short* vtb  = (unsigned short*)wsalloc((size_t)HD * NND * 2);  // V^T
  unsigned short* ob   = (unsigned short*)wsalloc((size_t)NND * HD * 2);
  unsigned short* tb   = (unsigned short*)wsalloc((size_t)NND * HD * 2);
  // pre-transposed bf16 weights
  unsigned short* wft = (unsigned short*)wsalloc((size_t)FD * HD * 2);
  unsigned short* wqt = (unsigned short*)wsalloc(LD * WSZ * 2);
  unsigned short* wkt = (unsigned short*)wsalloc(LD * WSZ * 2);
  unsigned short* wvt = (unsigned short*)wsalloc(LD * WSZ * 2);
  unsigned short* wot = (unsigned short*)wsalloc(LD * WSZ * 2);
  unsigned short* w1t = (unsigned short*)wsalloc(LD * WSZ * 2);
  unsigned short* w2t = (unsigned short*)wsalloc(LD * WSZ * 2);

  // one-time precompute
  k_tables<<<160, 256, 0, stream>>>(edge_emb, edge_dist_emb, W_ee, W_ed, tab);
  k_attn_bias<<<NND * NND / 256, 256, 0, stream>>>(ee, ed, tab, b_ee, b_ed, abias);
  k_f32_to_bf16<<<NND * HD / 256, 256, 0, stream>>>(x, xb, NND * HD);
  dim3 cgrid(WSZ / 256, LD);
  k_convt<<<dim3(FD * HD / 256, 1), 256, 0, stream>>>(W_feat, wft, FD, HD);
  k_convt<<<cgrid, 256, 0, stream>>>(Wq, wqt, HD, HD);
  k_convt<<<cgrid, 256, 0, stream>>>(Wk, wkt, HD, HD);
  k_convt<<<cgrid, 256, 0, stream>>>(Wv, wvt, HD, HD);
  k_convt<<<cgrid, 256, 0, stream>>>(Wo, wot, HD, HD);
  k_convt<<<cgrid, 256, 0, stream>>>(W1, w1t, HD, HD);
  k_convt<<<cgrid, 256, 0, stream>>>(W2, w2t, HD, HD);

  dim3 ggrid(NND / 64, HD / 64);   // 8 x 4 blocks of 64x64
  k_gemm_bf16<0, 0><<<ggrid, 128, 0, stream>>>(xb, wft, b_feat, 1.0f,
                                               hbuf, nullptr, NND, FD, HD);
  const float qscale = 0.17677669529663687f;   // 1/sqrt(32)
  for (int l = 0; l < LD; ++l) {
    k_layernorm<<<NND, HD, 0, stream>>>(hbuf, ln1_s + l * HD, ln1_b + l * HD, yb);
    k_gemm_bf16<0, 2><<<ggrid, 128, 0, stream>>>(yb, wqt + l * WSZ, bq + l * HD, qscale,
                                                 nullptr, qb, NND, HD, HD);
    k_gemm_bf16<0, 2><<<ggrid, 128, 0, stream>>>(yb, wkt + l * WSZ, bk + l * HD, 1.0f,
                                                 nullptr, kb16, NND, HD, HD);
    k_gemm_bf16<0, 3><<<ggrid, 128, 0, stream>>>(yb, wvt + l * WSZ, bv + l * HD, 1.0f,
                                                 nullptr, vtb, NND, HD, HD);
    k_attention<<<dim3(NHD, NND / 16), 32, 0, stream>>>(qb, kb16, vtb, abias, ob);
    k_gemm_bf16<0, 1><<<ggrid, 128, 0, stream>>>(ob, wot + l * WSZ, bo + l * HD, 1.0f,
                                                 hbuf, nullptr, NND, HD, HD);
    k_layernorm<<<NND, HD, 0, stream>>>(hbuf, ln2_s + l * HD, ln2_b + l * HD, yb);
    k_gemm_bf16<1, 2><<<ggrid, 128, 0, stream>>>(yb, w1t + l * WSZ, b1 + l * HD, 1.0f,
                                                 nullptr, tb, NND, HD, HD);
    k_gemm_bf16<0, 1><<<ggrid, 128, 0, stream>>>(tb, w2t + l * WSZ, b2 + l * HD, 1.0f,
                                                 hbuf, nullptr, NND, HD, HD);
  }
  (void)hipMemcpyAsync(d_out, hbuf, (size_t)NND * HD * sizeof(float),
                       hipMemcpyDeviceToDevice, stream);
}
